// _AFTBlock_30305289240684
// MI455X (gfx1250) — compile-verified, matrix operand order checked by reading, not run on hardware
//
#include <hip/hip_runtime.h>
#include <cstdint>
#include <cstddef>

// ---------------------------------------------------------------------------
// Problem: B=256, L=512, A=2048, W=64. Compute-bound (~70 GFLOP vs ~46 MB),
// so all GEMMs run on v_wmma_f32_16x16x32_bf16. All f32 operands are bulk-
// converted to bf16 once (memory-bound, ~4us), then GEMM kernels stage bf16
// tiles into double-buffered LDS via GLOBAL_LOAD_ASYNC_TO_LDS_B128 (ASYNCcnt)
// overlapping the next chunk's fill with the current chunk's WMMAs.
// ---------------------------------------------------------------------------
constexpr int NB = 256;
constexpr int NL = 512;
constexpr int NA = 2048;
constexpr int NW = 64;

constexpr int KC   = 32;  // k-chunk per WMMA step (bf16 16x16x32)
constexpr int LDST = 40;  // LDS row stride in bf16 elems (80B: 16B-aligned rows)

typedef __bf16 v16bf __attribute__((ext_vector_type(16)));
typedef float  v8f   __attribute__((ext_vector_type(8)));
typedef int    v4i_b __attribute__((vector_size(16)));   // matches builtin param

#if __has_builtin(__builtin_amdgcn_global_load_async_to_lds_b128)
#define HAS_ASYNC_LDS 1
#else
#define HAS_ASYNC_LDS 0
#endif

// ---------------------------------------------------------------------------
// Helpers
// ---------------------------------------------------------------------------
__device__ inline __bf16 f2bf(float x) {
    union { float f; uint32_t u; } v; v.f = x;
    uint32_t r = v.u + 0x7FFFu + ((v.u >> 16) & 1u);   // round-to-nearest-even
    uint16_t h = (uint16_t)(r >> 16);
    __bf16 out;
    __builtin_memcpy(&out, &h, 2);
    return out;
}

__device__ inline v8f zero8() {
    v8f z;
#pragma unroll
    for (int i = 0; i < 8; ++i) z[i] = 0.0f;
    return z;
}

__device__ inline v8f wmma_bf16(v16bf a, v16bf b, v8f c) {
    return __builtin_amdgcn_wmma_f32_16x16x32_bf16(
        /*neg_a=*/false, a, /*neg_b=*/false, b,
        /*c_mod=*/(short)0, c, /*reuse_a=*/false, /*reuse_b=*/false);
}

// 16-byte global->LDS copy. Async (memory->LDS direct, ASYNCcnt) if available.
__device__ inline void copy16(const __bf16* g, __bf16* l) {
#if HAS_ASYNC_LDS
    __builtin_amdgcn_global_load_async_to_lds_b128(
        (__attribute__((address_space(1))) v4i_b*)g,
        (__attribute__((address_space(3))) v4i_b*)l,
        /*offset=*/0, /*cpol=*/0);
#else
    *(uint4*)l = *(const uint4*)g;
#endif
}

__device__ inline void wait_async() {
#if HAS_ASYNC_LDS
#if __has_builtin(__builtin_amdgcn_s_wait_asynccnt)
    __builtin_amdgcn_s_wait_asynccnt(0);
#else
    asm volatile("s_wait_asynccnt 0" ::: "memory");
#endif
#endif
}

// Stage ROWS x 32 bf16 tile (row stride srcStride elems) into LDS.
// 16B chunks spread over 256 threads.
template<int ROWS>
__device__ inline void stage_tile(const __bf16* __restrict__ src, int srcStride,
                                  __bf16* dst, int tid) {
    constexpr int CH = ROWS * (KC / 8);   // 16B chunks (8 bf16 each)
    if constexpr (CH >= 256) {
#pragma unroll
        for (int it = 0; it < CH / 256; ++it) {
            int id = tid + it * 256;
            int r = id >> 2, c = (id & 3) << 3;
            copy16(src + (size_t)r * srcStride + c, dst + r * LDST + c);
        }
    } else {
        if (tid < CH) {
            int r = tid >> 2, c = (tid & 3) << 3;
            copy16(src + (size_t)r * srcStride + c, dst + r * LDST + c);
        }
    }
}

// Assemble one 16x32 bf16 WMMA operand fragment from an LDS tile.
// ISA 16-bit A/B layout: lane half 0 holds K={0..7,16..23}, half 1 holds
// K={8..15,24..31}; VGPR j holds the (even,odd) K pair.
__device__ inline v16bf load_frag(const __bf16* lds, int row, int half) {
    v16bf f;
    const __bf16* p = lds + row * LDST;
#pragma unroll
    for (int j = 0; j < 8; ++j) {
        int kk = ((j & 4) << 2) + (half << 3) + ((j & 3) << 1);
        f[2 * j]     = p[kk];
        f[2 * j + 1] = p[kk + 1];
    }
    return f;
}

__device__ inline void atomicMaxF(float* addr, float val) {
    int* ai = (int*)addr;
    int old = *ai;
    while (__int_as_float(old) < val) {
        int assumed = old;
        old = atomicCAS(ai, assumed, __float_as_int(val));
        if (old == assumed) break;
    }
}

// ---------------------------------------------------------------------------
// Kernel: bulk f32 -> bf16 conversion (one sweep per tensor)
// ---------------------------------------------------------------------------
__global__ __launch_bounds__(256) void cvt_bf16(const float* __restrict__ in,
                                                __bf16* __restrict__ out, int n4) {
    int i = blockIdx.x * blockDim.x + threadIdx.x;
    if (i < n4) {
        float4 v = ((const float4*)in)[i];
        __bf16 o[4] = { f2bf(v.x), f2bf(v.y), f2bf(v.z), f2bf(v.w) };
        uint2 p;
        __builtin_memcpy(&p, o, 8);
        ((uint2*)out)[i] = p;
    }
}

__global__ void aft_init_m(float* mArr) {
    int t = threadIdx.x;
    if (t < NW) mArr[t] = -__builtin_inff();
}

// ---------------------------------------------------------------------------
// Kernel: sigmoid(z_current @ Wq^T) -> sigQ [B,A] f32
// grid = (NA/64, NB/32), block 256 (8 wave32); 32x64 tile, 16x16 per wave.
// ---------------------------------------------------------------------------
__global__ __launch_bounds__(256) void aft_q(const __bf16* __restrict__ zc,
                                             const __bf16* __restrict__ Wq,
                                             float* __restrict__ sigQ) {
    __shared__ __bf16 sZ[2][32 * LDST];
    __shared__ __bf16 sW[2][64 * LDST];

    const int tid  = threadIdx.x;
    const int lane = tid & 31, wave = tid >> 5;
    const int half = lane >> 4, l16 = lane & 15;
    const int mi = wave >> 2, ni = wave & 3;
    const int aTile = blockIdx.x * 64;
    const int bTile = blockIdx.y * 32;

    auto issue = [&](int it, int buf) {
        int kc = it * KC;
        stage_tile<32>(zc + (size_t)bTile * NL + kc, NL, sZ[buf], tid);
        stage_tile<64>(Wq + (size_t)aTile * NL + kc, NL, sW[buf], tid);
    };

    constexpr int TOT = NL / KC;
    issue(0, 0);
    v8f acc = zero8();
    for (int it = 0; it < TOT; ++it) {
        int buf = it & 1;
        wait_async();
        __syncthreads();
        if (it + 1 < TOT) issue(it + 1, buf ^ 1);
        v16bf a = load_frag(sZ[buf], 16 * mi + l16, half);
        v16bf b = load_frag(sW[buf], 16 * ni + l16, half);
        acc = wmma_bf16(a, b, acc);
    }
    const int n = aTile + 16 * ni + l16;
#pragma unroll
    for (int r = 0; r < 8; ++r) {
        int m = bTile + 16 * mi + r + 8 * half;
        sigQ[(size_t)m * NA + n] =
            __builtin_amdgcn_rcpf(1.0f + __expf(-acc[r]));
    }
}

// ---------------------------------------------------------------------------
// Kernel: per-window scalar max of K_w = z_window[w] @ Wk^T
// grid = (NA/64, NB/32, NW), block 256
// ---------------------------------------------------------------------------
__global__ __launch_bounds__(256) void aft_max(const __bf16* __restrict__ zw,
                                               const __bf16* __restrict__ Wk,
                                               float* __restrict__ mArr) {
    __shared__ __bf16 sZ[2][32 * LDST];
    __shared__ __bf16 sW[2][64 * LDST];
    __shared__ float  red[256];

    const int tid  = threadIdx.x;
    const int lane = tid & 31, wave = tid >> 5;
    const int half = lane >> 4, l16 = lane & 15;
    const int mi = wave >> 2, ni = wave & 3;
    const int aTile = blockIdx.x * 64;
    const int bTile = blockIdx.y * 32;
    const int w     = blockIdx.z;

    const __bf16* zb = zw + ((size_t)w * NB + bTile) * NL;

    auto issue = [&](int it, int buf) {
        int kc = it * KC;
        stage_tile<32>(zb + kc, NL, sZ[buf], tid);
        stage_tile<64>(Wk + (size_t)aTile * NL + kc, NL, sW[buf], tid);
    };

    constexpr int TOT = NL / KC;
    issue(0, 0);
    v8f acc = zero8();
    for (int it = 0; it < TOT; ++it) {
        int buf = it & 1;
        wait_async();
        __syncthreads();
        if (it + 1 < TOT) issue(it + 1, buf ^ 1);
        v16bf a = load_frag(sZ[buf], 16 * mi + l16, half);
        v16bf b = load_frag(sW[buf], 16 * ni + l16, half);
        acc = wmma_bf16(a, b, acc);
    }
    float lm = acc[0];
#pragma unroll
    for (int r = 1; r < 8; ++r) lm = fmaxf(lm, acc[r]);
    red[tid] = lm;
    __syncthreads();
#pragma unroll
    for (int s = 128; s > 0; s >>= 1) {
        if (tid < s) red[tid] = fmaxf(red[tid], red[tid + s]);
        __syncthreads();
    }
    if (tid == 0) atomicMaxF(&mArr[w], red[0]);
}

// ---------------------------------------------------------------------------
// Kernel: main fused pass. Loop (w, kc) flattened & double-buffered:
// K,V tiles via WMMA; per-w epilogue e=exp(K-m[w]); num+=e*V; den+=e.
// Final: G = sigQ * num/(den+1e-8) as bf16.
// grid = (NA/64, NB/32), block 256
// ---------------------------------------------------------------------------
__global__ __launch_bounds__(256) void aft_main(const __bf16* __restrict__ zw,
                                                const __bf16* __restrict__ Wk,
                                                const __bf16* __restrict__ Wv,
                                                const float* __restrict__ mArr,
                                                const float* __restrict__ sigQ,
                                                __bf16* __restrict__ G) {
    __shared__ __bf16 sZ[2][32 * LDST];
    __shared__ __bf16 sK[2][64 * LDST];
    __shared__ __bf16 sV[2][64 * LDST];

    const int tid  = threadIdx.x;
    const int lane = tid & 31, wave = tid >> 5;
    const int half = lane >> 4, l16 = lane & 15;
    const int mi = wave >> 2, ni = wave & 3;
    const int aTile = blockIdx.x * 64;
    const int bTile = blockIdx.y * 32;

    auto issue = [&](int it, int buf) {
        int w  = it >> 4;            // NL/KC == 16 chunks per w
        int kc = (it & 15) * KC;
        stage_tile<32>(zw + ((size_t)w * NB + bTile) * NL + kc, NL, sZ[buf], tid);
        stage_tile<64>(Wk + (size_t)aTile * NL + kc, NL, sK[buf], tid);
        stage_tile<64>(Wv + (size_t)aTile * NL + kc, NL, sV[buf], tid);
    };

    float num[8], den[8];
#pragma unroll
    for (int r = 0; r < 8; ++r) { num[r] = 0.0f; den[r] = 0.0f; }

    constexpr int TOT = NW * (NL / KC);   // 1024
    issue(0, 0);
    v8f accK = zero8(), accV = zero8();
    for (int it = 0; it < TOT; ++it) {
        int buf = it & 1;
        wait_async();
        __syncthreads();
        if (it + 1 < TOT) issue(it + 1, buf ^ 1);

        v16bf a  = load_frag(sZ[buf], 16 * mi + l16, half);
        v16bf bk = load_frag(sK[buf], 16 * ni + l16, half);
        v16bf bv = load_frag(sV[buf], 16 * ni + l16, half);
        accK = wmma_bf16(a, bk, accK);
        accV = wmma_bf16(a, bv, accV);

        if ((it & 15) == 15) {            // end of one window step
            const float mw = mArr[it >> 4];
#pragma unroll
            for (int r = 0; r < 8; ++r) {
                float e = __expf(accK[r] - mw);
                num[r] += e * accV[r];
                den[r] += e;
            }
            accK = zero8();
            accV = zero8();
        }
    }

    const int n = aTile + 16 * ni + l16;
#pragma unroll
    for (int r = 0; r < 8; ++r) {
        int m = bTile + 16 * mi + r + 8 * half;
        float ctx = num[r] * __builtin_amdgcn_rcpf(den[r] + 1e-8f);
        G[(size_t)m * NA + n] = f2bf(sigQ[(size_t)m * NA + n] * ctx);
    }
}

// ---------------------------------------------------------------------------
// Kernel: out = G @ Wo^T   [B,A]x[A,L] -> [B,L]
// grid = (NL/64, NB/32), block 256
// ---------------------------------------------------------------------------
__global__ __launch_bounds__(256) void aft_out(const __bf16* __restrict__ G,
                                               const __bf16* __restrict__ Wo,
                                               float* __restrict__ out) {
    __shared__ __bf16 sG[2][32 * LDST];
    __shared__ __bf16 sW[2][64 * LDST];

    const int tid  = threadIdx.x;
    const int lane = tid & 31, wave = tid >> 5;
    const int half = lane >> 4, l16 = lane & 15;
    const int mi = wave >> 2, ni = wave & 3;
    const int lTile = blockIdx.x * 64;
    const int bTile = blockIdx.y * 32;

    auto issue = [&](int it, int buf) {
        int kc = it * KC;
        stage_tile<32>(G  + (size_t)bTile * NA + kc, NA, sG[buf], tid);
        stage_tile<64>(Wo + (size_t)lTile * NA + kc, NA, sW[buf], tid);
    };

    constexpr int TOT = NA / KC;          // 64
    issue(0, 0);
    v8f acc = zero8();
    for (int it = 0; it < TOT; ++it) {
        int buf = it & 1;
        wait_async();
        __syncthreads();
        if (it + 1 < TOT) issue(it + 1, buf ^ 1);
        v16bf a = load_frag(sG[buf], 16 * mi + l16, half);
        v16bf b = load_frag(sW[buf], 16 * ni + l16, half);
        acc = wmma_bf16(a, b, acc);
    }
    const int n = lTile + 16 * ni + l16;
#pragma unroll
    for (int r = 0; r < 8; ++r) {
        int m = bTile + 16 * mi + r + 8 * half;
        out[(size_t)m * NL + n] = acc[r];
    }
}

// ---------------------------------------------------------------------------
// Launch
// ---------------------------------------------------------------------------
extern "C" void kernel_launch(void* const* d_in, const int* in_sizes, int n_in,
                              void* d_out, int out_size, void* d_ws, size_t ws_size,
                              hipStream_t stream) {
    const float* zc = (const float*)d_in[0];   // [B,L]
    const float* zw = (const float*)d_in[1];   // [W,B,L]
    const float* Wq = (const float*)d_in[2];   // [A,L]
    const float* Wk = (const float*)d_in[3];   // [A,L]
    const float* Wv = (const float*)d_in[4];   // [A,L]
    const float* Wo = (const float*)d_in[5];   // [L,A]
    float* out = (float*)d_out;                // [B,L]

    char* ws = (char*)d_ws;
    size_t off = 0;
    float*  mArr = (float*)(ws + off);  off += 256;
    float*  sigQ = (float*)(ws + off);  off += (size_t)NB * NA * 4;   // 2 MB
    __bf16* G    = (__bf16*)(ws + off); off += (size_t)NB * NA * 2;   // 1 MB
    __bf16* zc16 = (__bf16*)(ws + off); off += (size_t)NB * NL * 2;
    __bf16* zw16 = (__bf16*)(ws + off); off += (size_t)NW * NB * NL * 2; // 16 MB
    __bf16* Wq16 = (__bf16*)(ws + off); off += (size_t)NA * NL * 2;
    __bf16* Wk16 = (__bf16*)(ws + off); off += (size_t)NA * NL * 2;
    __bf16* Wv16 = (__bf16*)(ws + off); off += (size_t)NA * NL * 2;
    __bf16* Wo16 = (__bf16*)(ws + off); off += (size_t)NL * NA * 2;

    auto cvt = [&](const float* in, __bf16* o, size_t n) {
        int n4 = (int)(n / 4);
        cvt_bf16<<<(n4 + 255) / 256, 256, 0, stream>>>(in, o, n4);
    };
    cvt(zc, zc16, (size_t)NB * NL);
    cvt(zw, zw16, (size_t)NW * NB * NL);
    cvt(Wq, Wq16, (size_t)NA * NL);
    cvt(Wk, Wk16, (size_t)NA * NL);
    cvt(Wv, Wv16, (size_t)NA * NL);
    cvt(Wo, Wo16, (size_t)NL * NA);

    aft_init_m<<<1, 64, 0, stream>>>(mArr);
    aft_q   <<<dim3(NA / 64, NB / 32),     256, 0, stream>>>(zc16, Wq16, sigQ);
    aft_max <<<dim3(NA / 64, NB / 32, NW), 256, 0, stream>>>(zw16, Wk16, mArr);
    aft_main<<<dim3(NA / 64, NB / 32),     256, 0, stream>>>(zw16, Wk16, Wv16, mArr, sigQ, G);
    aft_out <<<dim3(NL / 64, NB / 32),     256, 0, stream>>>(G, Wo16, out);
}
